// Model_TEM4CTR_82325933129930
// MI455X (gfx1250) — matrix operand order repaired and checked by reading, not compiled, not run on hardware
//
#include <hip/hip_runtime.h>

typedef __attribute__((ext_vector_type(16))) _Float16 v16h;
typedef __attribute__((ext_vector_type(8)))  _Float16 v8h;
typedef __attribute__((ext_vector_type(8)))  float    v8f;

constexpr int NB = 64;    // batch
constexpr int NS = 256;   // seq
constexpr int NK = 10;    // unclicked per clicked
constexpr int ND = 64;    // dim
constexpr float BIG_NEG = -4294967295.0f;

__device__ __forceinline__ v16h cat8(v8h lo, v8h hi) {
  return __builtin_shufflevector(lo, hi, 0,1,2,3,4,5,6,7,8,9,10,11,12,13,14,15);
}
__device__ __forceinline__ float sigmoidf(float x) { return 1.0f / (1.0f + __expf(-x)); }

// ---------------------------------------------------------------------------
// Pack a row-major f32 weight matrix W[Kdim][N] into WMMA-B per-lane layout:
// out[kt][nt][lane][h] (f16), where col = nt*16 + (lane&15),
// kk = kt*32 + (lane>>4)*8 + (h&7) + (h>>3)*16. Zero-padded outside (Kdim, N).
// ---------------------------------------------------------------------------
__global__ void pack_w_kernel(const float* __restrict__ W, int Kdim, int N,
                              int ntiles, int total, _Float16* __restrict__ out) {
  int i = blockIdx.x * blockDim.x + threadIdx.x;
  if (i >= total) return;
  int h    = i & 15;
  int lane = (i >> 4) & 31;
  int t    = i >> 9;
  int nt   = t % ntiles;
  int kt   = t / ntiles;
  int col  = nt * 16 + (lane & 15);
  int kk   = kt * 32 + (lane >> 4) * 8 + (h & 7) + (h >> 3) * 16;
  float v  = (kk < Kdim && col < N) ? W[(size_t)kk * N + col] : 0.0f;
  out[i] = (_Float16)v;
}

// ---------------------------------------------------------------------------
// Gather item = emb[mid], click = emb[click_his] * click_mask
// ---------------------------------------------------------------------------
__global__ void gather_kernel(const int* __restrict__ mid,
                              const int* __restrict__ chis,
                              const float* __restrict__ cmask,
                              const float* __restrict__ emb,
                              float* __restrict__ item,
                              float* __restrict__ click) {
  int i0 = blockIdx.x * blockDim.x + threadIdx.x;
  if (i0 < NB * ND) item[i0] = emb[(size_t)mid[i0 >> 6] * ND + (i0 & 63)];
  for (int j = i0; j < NB * NS * ND; j += gridDim.x * blockDim.x) {
    int bs = j >> 6;
    click[j] = emb[(size_t)chis[bs] * ND + (j & 63)] * cmask[bs];
  }
}

// ---------------------------------------------------------------------------
// Fused DIN MLP scorer. One wave32 handles a 16-row M-tile of the MLP
//   x[256] -> sigmoid(x@W1+b1)[80] -> sigmoid(@W2+b2)[40] -> @W3+b3 -> score
// Layers 1&2 run on V_WMMA_F32_16X16X32_F16; layer 3 is a 40-wide VALU dot.
// mode 0: rows r = (b,s,k) over B*S*K; q = click[b,s], fact = emb[unclick]*umask
// mode 1: rows r = (b,s)  over B*S;    q = item[b],    fact = fbase[r]
// ---------------------------------------------------------------------------
__global__ __launch_bounds__(128) void din_scores_kernel(
    int mode, int mtiles,
    const float* __restrict__ qbase,
    const float* __restrict__ fbase,
    const int*   __restrict__ his,
    const float* __restrict__ hmask,
    const float* __restrict__ emb,
    const float* __restrict__ smask,     // score mask, indexed by flat row
    const _Float16* __restrict__ w1p,    // [8][5][32][16] f16
    const float* __restrict__ b1,        // [80]
    const _Float16* __restrict__ w2p,    // [3][3][32][16] f16
    const float* __restrict__ b2,        // [40]
    const float* __restrict__ w3,        // [40]
    const float* __restrict__ b3,        // [1]
    float* __restrict__ scores)
{
  __shared__ __align__(16) _Float16 xs [4][16][256]; // 8 KB / wave
  __shared__ __align__(16) _Float16 hs1[4][16][96];  // 3 KB / wave (K-padded)
  __shared__ __align__(16) float    hs2[4][16][48];  // 3 KB / wave

  const int wid  = threadIdx.x >> 5;
  const int lane = threadIdx.x & 31;
  const int tile = blockIdx.x * 4 + wid;
  if (tile >= mtiles) return;                 // wave-uniform: EXEC all-1 below
  const int row0 = tile * 16;
  const int arow = lane & 15;                 // A-matrix row / B column lane
  const int agrp = (lane >> 4) * 8;           // A K-offset & C/D row offset

  // ---- build x = [q, f, q-f, q*f] tile in LDS (f16, row-major 16x256) ----
  for (int r = 0; r < 16; ++r) {
    const int rg = row0 + r;
    const float* qp; const float* fp; float fs;
    if (mode == 0) {
      int b = rg / (NS * NK); int rem = rg - b * (NS * NK);
      int s = rem / NK;       int k   = rem - s * NK;
      qp = qbase + (size_t)(b * NS + s) * ND;
      int hidx = b * (NK * NS) + s * NK + k;
      fp = emb + (size_t)his[hidx] * ND;
      fs = hmask[hidx];
    } else {
      int b = rg / NS;
      qp = qbase + (size_t)b * ND;
      fp = fbase + (size_t)rg * ND;
      fs = 1.0f;
    }
    __builtin_prefetch(fp, 0, 0);
#pragma unroll
    for (int j = 0; j < 2; ++j) {
      int d = lane + j * 32;
      float q = qp[d];
      float f = fp[d] * fs;
      xs[wid][r][d]       = (_Float16)q;
      xs[wid][r][64 + d]  = (_Float16)f;
      xs[wid][r][128 + d] = (_Float16)(q - f);
      xs[wid][r][192 + d] = (_Float16)(q * f);
    }
  }

  // ---- layer 1: 16x256 @ 256x80, 8 K-steps x 5 N-tiles ----
  v8f acc[5] = {};
#pragma unroll
  for (int kt = 0; kt < 8; ++kt) {
    const _Float16* ap = &xs[wid][arow][kt * 32 + agrp];
    v16h a = cat8(*(const v8h*)ap, *(const v8h*)(ap + 16));
#pragma unroll
    for (int nt = 0; nt < 5; ++nt) {
      v16h bm = *(const v16h*)(w1p + (size_t)((kt * 5 + nt) * 32 + lane) * 16);
      acc[nt] = __builtin_amdgcn_wmma_f32_16x16x32_f16(
          false, a, false, bm, (short)0, acc[nt], false, false);
    }
  }

  // sigmoid(+bias) -> LDS as layer-2 A operand (cols 80..95 zero-padded)
#pragma unroll
  for (int nt = 0; nt < 5; ++nt) {
    int col = nt * 16 + arow;
    float bias = b1[col];
#pragma unroll
    for (int i = 0; i < 8; ++i)
      hs1[wid][agrp + i][col] = (_Float16)sigmoidf(acc[nt][i] + bias);
  }
  for (int i = lane; i < 256; i += 32)
    hs1[wid][i >> 4][80 + (i & 15)] = (_Float16)0.0f;

  // ---- layer 2: 16x96(pad) @ 96x48(pad), 3 K-steps x 3 N-tiles ----
  v8f acc2[3] = {};
#pragma unroll
  for (int kt = 0; kt < 3; ++kt) {
    const _Float16* ap = &hs1[wid][arow][kt * 32 + agrp];
    v16h a = cat8(*(const v8h*)ap, *(const v8h*)(ap + 16));
#pragma unroll
    for (int nt = 0; nt < 3; ++nt) {
      v16h bm = *(const v16h*)(w2p + (size_t)((kt * 3 + nt) * 32 + lane) * 16);
      acc2[nt] = __builtin_amdgcn_wmma_f32_16x16x32_f16(
          false, a, false, bm, (short)0, acc2[nt], false, false);
    }
  }

#pragma unroll
  for (int nt = 0; nt < 3; ++nt) {
    int col = nt * 16 + arow;
    float bias = (col < 40) ? b2[col] : 0.0f;
#pragma unroll
    for (int i = 0; i < 8; ++i)
      hs2[wid][agrp + i][col] = sigmoidf(acc2[nt][i] + bias);
  }

  // ---- layer 3 + score mask ----
  if (lane < 16) {
    float sum = b3[0];
    for (int c = 0; c < 40; ++c) sum += hs2[wid][lane][c] * w3[c];
    int rg = row0 + lane;
    float mv = smask[rg];
    scores[rg] = (mv == 1.0f) ? sum : BIG_NEG;
  }
}

// ---------------------------------------------------------------------------
// din1 epilogue per (b,s): softmax over K, un_att = sum_k a_k * fact_k,
// denoise: facts2 = click * (1 + (un_att.click)/(||click||^2 + 1e-12))
// One wave32 per (b,s); lanes split D=64.
// ---------------------------------------------------------------------------
__global__ __launch_bounds__(128) void din1_combine_kernel(
    const float* __restrict__ scores1, const int* __restrict__ uhis,
    const float* __restrict__ umask,   const float* __restrict__ emb,
    const float* __restrict__ click,
    float* __restrict__ un_att, float* __restrict__ facts2)
{
  int wid = threadIdx.x >> 5, lane = threadIdx.x & 31;
  int bs = blockIdx.x * 4 + wid;
  if (bs >= NB * NS) return;
  int b = bs / NS, s = bs - b * NS;
  const float* sc = scores1 + (size_t)bs * NK;

  float mx = -3.4e38f;
#pragma unroll
  for (int k = 0; k < NK; ++k) mx = fmaxf(mx, sc[k]);
  float e[NK]; float se = 0.f;
#pragma unroll
  for (int k = 0; k < NK; ++k) { e[k] = __expf(sc[k] - mx); se += e[k]; }
  float inv = 1.0f / se;

  int hbase = b * (NK * NS) + s * NK;
  int idx[NK]; float w[NK];
#pragma unroll
  for (int k = 0; k < NK; ++k) {
    idx[k] = uhis[hbase + k];
    w[k]   = e[k] * inv * umask[hbase + k];   // fold fact-mask into weight
  }

  float dp = 0.f, nq = 0.f;
  float cl[2];
#pragma unroll
  for (int j = 0; j < 2; ++j) {
    int d = lane + j * 32;
    float a = 0.f;
#pragma unroll
    for (int k = 0; k < NK; ++k) a += w[k] * emb[(size_t)idx[k] * ND + d];
    float c = click[(size_t)bs * ND + d];
    cl[j] = c;
    un_att[(size_t)bs * ND + d] = a;
    dp += a * c; nq += c * c;
  }
#pragma unroll
  for (int off = 16; off > 0; off >>= 1) {
    dp += __shfl_xor(dp, off);
    nq += __shfl_xor(nq, off);
  }
  float ratio = dp / (nq + 1e-12f);
#pragma unroll
  for (int j = 0; j < 2; ++j) {
    int d = lane + j * 32;
    facts2[(size_t)bs * ND + d] = cl[j] * (1.0f + ratio);
  }
}

// ---------------------------------------------------------------------------
// Attention pooling over S=256: one block per b; softmax + weighted sum.
// ---------------------------------------------------------------------------
__global__ __launch_bounds__(256) void attn_pool_kernel(
    const float* __restrict__ scores, const float* __restrict__ facts,
    float* __restrict__ att)
{
  __shared__ float red[256];
  __shared__ float wgt[256];
  int b = blockIdx.x, t = threadIdx.x;
  float sc = scores[(size_t)b * NS + t];
  red[t] = sc; __syncthreads();
  for (int o = 128; o > 0; o >>= 1) {
    if (t < o) red[t] = fmaxf(red[t], red[t + o]);
    __syncthreads();
  }
  float mx = red[0]; __syncthreads();
  float e = __expf(sc - mx);
  red[t] = e; __syncthreads();
  for (int o = 128; o > 0; o >>= 1) {
    if (t < o) red[t] += red[t + o];
    __syncthreads();
  }
  float se = red[0];
  wgt[t] = e / se; __syncthreads();
  if (t < ND) {
    float a = 0.f;
    for (int s = 0; s < NS; ++s)
      a += wgt[s] * facts[((size_t)b * NS + s) * ND + t];
    att[(size_t)b * ND + t] = a;
  }
}

// ---------------------------------------------------------------------------
// Final head: concat + inference batchnorm; two PReLU FC layers; softmax out.
// ---------------------------------------------------------------------------
__global__ void head_bn_kernel(const float* __restrict__ item,
                               const float* __restrict__ att1,
                               const float* __restrict__ att2,
                               const float* __restrict__ g, const float* __restrict__ bb,
                               const float* __restrict__ m, const float* __restrict__ v,
                               float* __restrict__ xh) {
  int i = blockIdx.x * blockDim.x + threadIdx.x;
  if (i >= NB * 192) return;
  int b = i / 192, c = i - b * 192;
  float x = (c < 64) ? item[b * 64 + c]
          : (c < 128) ? att1[b * 64 + c - 64]
                      : att2[b * 64 + c - 128];
  x = (x - m[c]) / sqrtf(v[c] + 1e-3f) * g[c] + bb[c];
  xh[i] = x;
}

__global__ void head_fc_kernel(const float* __restrict__ in, const float* __restrict__ W,
                               const float* __restrict__ bias, const float* __restrict__ alpha,
                               float* __restrict__ out, int In, int On) {
  int i = blockIdx.x * blockDim.x + threadIdx.x;
  if (i >= NB * On) return;
  int b = i / On, c = i - b * On;
  float s = bias[c];
  for (int j = 0; j < In; ++j) s += in[(size_t)b * In + j] * W[(size_t)j * On + c];
  out[i] = fmaxf(s, 0.f) + alpha[c] * fminf(s, 0.f);
}

__global__ void head_out_kernel(const float* __restrict__ h2, const float* __restrict__ W,
                                const float* __restrict__ bias, float* __restrict__ out) {
  int b = threadIdx.x;
  if (b >= NB) return;
  float l0 = bias[0], l1 = bias[1];
  for (int j = 0; j < 80; ++j) {
    float h = h2[b * 80 + j];
    l0 += h * W[j * 2];
    l1 += h * W[j * 2 + 1];
  }
  float mx = fmaxf(l0, l1);
  float e0 = __expf(l0 - mx), e1 = __expf(l1 - mx);
  float se = e0 + e1;
  out[b * 2]     = e0 / se + 1e-8f;
  out[b * 2 + 1] = e1 / se + 1e-8f;
}

// ---------------------------------------------------------------------------
extern "C" void kernel_launch(void* const* d_in, const int* in_sizes, int n_in,
                              void* d_out, int out_size, void* d_ws, size_t ws_size,
                              hipStream_t stream) {
  (void)in_sizes; (void)n_in; (void)out_size; (void)ws_size;

  const int*   mid          = (const int*)  d_in[0];
  const int*   click_his    = (const int*)  d_in[1];
  const int*   unclick_his  = (const int*)  d_in[2];
  const float* click_mask   = (const float*)d_in[3];
  const float* unclick_mask = (const float*)d_in[4];
  const float* emb          = (const float*)d_in[5];
  // attention MLPs: s (din1), c (att1), u (att2)
  const float* mW1[3] = {(const float*)d_in[6],  (const float*)d_in[12], (const float*)d_in[18]};
  const float* mb1[3] = {(const float*)d_in[7],  (const float*)d_in[13], (const float*)d_in[19]};
  const float* mW2[3] = {(const float*)d_in[8],  (const float*)d_in[14], (const float*)d_in[20]};
  const float* mb2[3] = {(const float*)d_in[9],  (const float*)d_in[15], (const float*)d_in[21]};
  const float* mW3[3] = {(const float*)d_in[10], (const float*)d_in[16], (const float*)d_in[22]};
  const float* mb3[3] = {(const float*)d_in[11], (const float*)d_in[17], (const float*)d_in[23]};
  const float* bn_g = (const float*)d_in[24];
  const float* bn_b = (const float*)d_in[25];
  const float* bn_m = (const float*)d_in[26];
  const float* bn_v = (const float*)d_in[27];
  const float* fW1 = (const float*)d_in[28];
  const float* fb1 = (const float*)d_in[29];
  const float* a1  = (const float*)d_in[30];
  const float* fW2 = (const float*)d_in[31];
  const float* fb2 = (const float*)d_in[32];
  const float* a2  = (const float*)d_in[33];
  const float* fW3 = (const float*)d_in[34];
  const float* fb3 = (const float*)d_in[35];

  // ---- workspace partition (~14 MB) ----
  char* p = (char*)d_ws;
  auto alloc = [&](size_t bytes) -> void* {
    void* r = (void*)p;
    p += (bytes + 255) & ~(size_t)255;
    return r;
  };
  _Float16* w1p[3]; _Float16* w2p[3];
  for (int i = 0; i < 3; ++i) {
    w1p[i] = (_Float16*)alloc((size_t)8 * 5 * 512 * 2);
    w2p[i] = (_Float16*)alloc((size_t)3 * 3 * 512 * 2);
  }
  float* item    = (float*)alloc((size_t)NB * ND * 4);
  float* click   = (float*)alloc((size_t)NB * NS * ND * 4);
  float* un_att  = (float*)alloc((size_t)NB * NS * ND * 4);
  float* facts2  = (float*)alloc((size_t)NB * NS * ND * 4);
  float* scores1 = (float*)alloc((size_t)NB * NS * NK * 4);
  float* scores2 = (float*)alloc((size_t)NB * NS * 4);
  float* scores3 = (float*)alloc((size_t)NB * NS * 4);
  float* att1    = (float*)alloc((size_t)NB * ND * 4);
  float* att2    = (float*)alloc((size_t)NB * ND * 4);
  float* xh      = (float*)alloc((size_t)NB * 192 * 4);
  float* h1      = (float*)alloc((size_t)NB * 200 * 4);
  float* h2      = (float*)alloc((size_t)NB * 80 * 4);

  // ---- pack attention-MLP weights into WMMA-B layout (f16) ----
  for (int m = 0; m < 3; ++m) {
    pack_w_kernel<<<(8 * 5 * 512 + 255) / 256, 256, 0, stream>>>(
        mW1[m], 256, 80, 5, 8 * 5 * 512, w1p[m]);
    pack_w_kernel<<<(3 * 3 * 512 + 255) / 256, 256, 0, stream>>>(
        mW2[m], 80, 40, 3, 3 * 3 * 512, w2p[m]);
  }

  // ---- gather item & masked click history ----
  gather_kernel<<<4096, 256, 0, stream>>>(mid, click_his, click_mask, emb, item, click);

  // ---- din #1 scores: M = B*S*K = 163840 rows -> 10240 tiles ----
  din_scores_kernel<<<2560, 128, 0, stream>>>(
      0, 10240, click, nullptr, unclick_his, unclick_mask, emb, unclick_mask,
      w1p[0], mb1[0], w2p[0], mb2[0], mW3[0], mb3[0], scores1);

  // ---- din #1 softmax/weighted-sum + denoise projection ----
  din1_combine_kernel<<<4096, 128, 0, stream>>>(
      scores1, unclick_his, unclick_mask, emb, click, un_att, facts2);

  // ---- din #2 (c-weights, facts = click+shared) and #3 (u-weights, facts = un_att) ----
  din_scores_kernel<<<256, 128, 0, stream>>>(
      1, 1024, item, facts2, nullptr, nullptr, nullptr, click_mask,
      w1p[1], mb1[1], w2p[1], mb2[1], mW3[1], mb3[1], scores2);
  din_scores_kernel<<<256, 128, 0, stream>>>(
      1, 1024, item, un_att, nullptr, nullptr, nullptr, click_mask,
      w1p[2], mb1[2], w2p[2], mb2[2], mW3[2], mb3[2], scores3);

  // ---- attention pooling over S ----
  attn_pool_kernel<<<NB, 256, 0, stream>>>(scores2, facts2, att1);
  attn_pool_kernel<<<NB, 256, 0, stream>>>(scores3, un_att, att2);

  // ---- final head ----
  head_bn_kernel<<<(NB * 192 + 255) / 256, 256, 0, stream>>>(
      item, att1, att2, bn_g, bn_b, bn_m, bn_v, xh);
  head_fc_kernel<<<(NB * 200 + 255) / 256, 256, 0, stream>>>(xh, fW1, fb1, a1, h1, 192, 200);
  head_fc_kernel<<<(NB * 80 + 255) / 256, 256, 0, stream>>>(h1, fW2, fb2, a2, h2, 200, 80);
  head_out_kernel<<<1, 64, 0, stream>>>(h2, fW3, fb3, (float*)d_out);
}
